// RNNTJoiner_4698694222681
// MI455X (gfx1250) — compile-verified, matrix operand order checked by reading, not run on hardware
//
#include <hip/hip_runtime.h>

typedef __attribute__((ext_vector_type(16))) _Float16 v16h;
typedef __attribute__((ext_vector_type(8)))  _Float16 v8h;
typedef __attribute__((ext_vector_type(4)))  _Float16 v4h;
typedef __attribute__((ext_vector_type(8)))  float    v8f;
typedef __attribute__((ext_vector_type(4)))  unsigned int u32x4;
typedef __attribute__((ext_vector_type(4)))  int          i32x4;
typedef __attribute__((ext_vector_type(8)))  int          i32x8;

namespace {
constexpr int kB = 8, kT = 128, kU = 64, kD = 1024, kV = 1024;
constexpr int BM = 128, BN = 128, BK = 32;
constexpr int NK  = kD / BK;      // 32 K-steps
constexpr int LDA = BK + 8;       // 40 halves/row: 80B stride -> conflict-free b128 reads
constexpr int LDB = BK + 8;       // TDM pad: 16 DW data + 4 DW pad = 80B row stride
}

// ---------------------------------------------------------------------------
// Pre-pass: W[k][v] fp32  ->  Wt[v][k] f16  (LDS-tiled 64x64 transpose+convert)
// ---------------------------------------------------------------------------
__global__ __launch_bounds__(256)
void transpose_w_f16(const float* __restrict__ W, _Float16* __restrict__ Wt) {
  __shared__ float tile[64][65];
  const int v0 = blockIdx.x * 64;
  const int k0 = blockIdx.y * 64;
  const int tx = threadIdx.x & 63;   // inner (coalesced) index
  const int ty = threadIdx.x >> 6;   // 0..3
#pragma unroll
  for (int p = 0; p < 16; ++p) {
    const int k = ty + p * 4;
    tile[k][tx] = W[(size_t)(k0 + k) * kV + v0 + tx];
  }
  __syncthreads();
#pragma unroll
  for (int p = 0; p < 16; ++p) {
    const int v = ty + p * 4;
    Wt[(size_t)(v0 + v) * kD + k0 + tx] = (_Float16)tile[tx][v];
  }
}

// ---------------------------------------------------------------------------
// Main kernel: A produced on the fly (relu(src+tgt) -> f16 -> LDS),
// W tile DMA'd by the Tensor Data Mover straight into padded LDS layout.
// ---------------------------------------------------------------------------
__global__ __launch_bounds__(256, 1)
void rnnt_joiner_wmma_tdm(const float* __restrict__ src,
                          const float* __restrict__ tgt,
                          const _Float16* __restrict__ Wt,   // [v][k] f16
                          const float* __restrict__ bias,
                          float* __restrict__ out)
{
  __shared__ _Float16 sA[2][BM * LDA];   // A tile: [row m][k]
  __shared__ _Float16 sB[2][BN * LDB];   // W tile: [col v][k], TDM-padded rows

  const int tid = threadIdx.x;
  const int bn0 = blockIdx.x * BN;
  const int bm0 = blockIdx.y * BM;

  // ---- producer mapping: A tile (128 rows x 32 k fp32), float4/thread x 4 passes
  const int a_c = (tid & 7) * 4;
  const int a_r = tid >> 3;
  int aSrcOff[4], aTgtOff[4], aRow[4];
#pragma unroll
  for (int p = 0; p < 4; ++p) {
    const int r    = a_r + p * 32;
    const int m    = bm0 + r;
    const int u    = m & (kU - 1);
    const int pidx = m >> 6;             // b*T + t
    const int b    = pidx >> 7;          // / T
    aRow[p]    = r;
    aSrcOff[p] = pidx * kD;
    aTgtOff[p] = (b * kU + u) * kD;
  }

  float4 ra[4];
  auto load_a = [&](int ks) {
    const int k0 = ks * BK;
#pragma unroll
    for (int p = 0; p < 4; ++p) {
      const float4 s = *(const float4*)(src + aSrcOff[p] + k0 + a_c);
      const float4 t = *(const float4*)(tgt + aTgtOff[p] + k0 + a_c);
      float4 j;
      j.x = fmaxf(s.x + t.x, 0.f);
      j.y = fmaxf(s.y + t.y, 0.f);
      j.z = fmaxf(s.z + t.z, 0.f);
      j.w = fmaxf(s.w + t.w, 0.f);
      ra[p] = j;
    }
  };
  auto store_a = [&](int buf) {
#pragma unroll
    for (int p = 0; p < 4; ++p) {
      v4h h;
      h[0] = (_Float16)ra[p].x; h[1] = (_Float16)ra[p].y;
      h[2] = (_Float16)ra[p].z; h[3] = (_Float16)ra[p].w;
      *(v4h*)(&sA[buf][aRow[p] * LDA + a_c]) = h;
    }
  };

  // ---- TDM descriptor group 1 (constant): 2D tile 32(k) x 128(v), f16,
  //      pad 4 DW after every 16 DW -> 80B LDS row stride.
  i32x8 g1;
  g1[0] = (1 << 16)            // data_size = 2 bytes
        | (1 << 20)            // pad_enable
        | (3 << 22)            // pad_interval: 16 dwords
        | (3 << 25);           // pad_amount:   4 dwords
  g1[1] = (int)(32u << 16);    // tensor_dim0 = 32 (lo16 in [31:16])
  g1[2] = (int)(128u << 16);   // tensor_dim0 hi = 0; tensor_dim1 = 128 (lo16)
  g1[3] = (int)(32u << 16);    // tensor_dim1 hi = 0; tile_dim0 = 32
  g1[4] = 128;                 // tile_dim1 = 128; tile_dim2 = 0
  g1[5] = (int)(unsigned)kD;   // tensor_dim0_stride = 1024 (lo32)
  g1[6] = 0;
  g1[7] = 0;
  const i32x4 z4 = {0, 0, 0, 0};

  const int w = tid >> 5;
  auto tdm_issue = [&](int ks, int buf) {
#if defined(__HIP_DEVICE_COMPILE__)
    const unsigned lds_off = (unsigned)(size_t)(&sB[buf][0]);
    const unsigned long long ga =
        (unsigned long long)(size_t)(Wt + (size_t)bn0 * kD + ks * BK);
    u32x4 g0;
    g0[0] = 1u;                                        // count=1 user D#
    g0[1] = lds_off;                                   // lds_addr
    g0[2] = (unsigned)(ga & 0xffffffffu);              // global_addr lo
    g0[3] = (unsigned)((ga >> 32) & 0x01ffffffu)       // global_addr hi (57b)
          | (2u << 30);                                // type = 2 ("image")
#if __has_include(<hip/amd_detail/amd_gfx1250_TDM.h>)
    const i32x8 z8 = {0, 0, 0, 0, 0, 0, 0, 0};
    __builtin_amdgcn_tensor_load_to_lds(g0, g1, z4, z4, z8, 0);   // 6-arg toolchain
#else
    __builtin_amdgcn_tensor_load_to_lds(g0, g1, z4, z4, 0);       // 5-arg (ROCm 7.2)
#endif
#endif
  };

  // ---- consumer mapping (wave32): 8 waves as 4(M) x 2(N); each wave 32x64 of C
  const int lane  = tid & 31;
  const int wm    = (w & 3) * 32;
  const int wn    = (w >> 2) * 64;
  const int lrow  = lane & 15;
  const int khalf = (lane >> 4) * 8;

  v8f acc[2][4] = {};

  if (w == 0) tdm_issue(0, 0);
  load_a(0);
#pragma unroll 2
  for (int ks = 0; ks < NK; ++ks) {
    const int buf = ks & 1;
    store_a(buf);
#if defined(__HIP_DEVICE_COMPILE__)
    if (w == 0) __builtin_amdgcn_s_wait_tensorcnt(0);  // W tile for `buf` landed
#endif
    __syncthreads();
    if (ks + 1 < NK) {
      load_a(ks + 1);                       // overlap global latency with WMMAs
      if (w == 0) tdm_issue(ks + 1, buf ^ 1);  // overlap DMA with WMMAs
    }

    v16h afr[2], bfr[4];
#pragma unroll
    for (int i = 0; i < 2; ++i) {
      const _Float16* pa = &sA[buf][(wm + i * 16 + lrow) * LDA + khalf];
      const v8h lo = *(const v8h*)pa;
      const v8h hi = *(const v8h*)(pa + 16);
#pragma unroll
      for (int e = 0; e < 8; ++e) { afr[i][e] = lo[e]; afr[i][e + 8] = hi[e]; }
    }
#pragma unroll
    for (int j = 0; j < 4; ++j) {
      const _Float16* pb = &sB[buf][(wn + j * 16 + lrow) * LDB + khalf];
      const v8h lo = *(const v8h*)pb;
      const v8h hi = *(const v8h*)(pb + 16);
#pragma unroll
      for (int e = 0; e < 8; ++e) { bfr[j][e] = lo[e]; bfr[j][e + 8] = hi[e]; }
    }
#pragma unroll
    for (int i = 0; i < 2; ++i)
#pragma unroll
      for (int j = 0; j < 4; ++j)
        acc[i][j] = __builtin_amdgcn_wmma_f32_16x16x32_f16(
            false, afr[i], false, bfr[j], (short)0, acc[i][j], false, false);
  }

#pragma unroll
  for (int j = 0; j < 4; ++j) {
    const int col = bn0 + wn + j * 16 + lrow;
    const float bv = bias[col];
#pragma unroll
    for (int i = 0; i < 2; ++i) {
      const int r0 = bm0 + wm + i * 16 + ((lane >> 4) << 3);
#pragma unroll
      for (int g = 0; g < 8; ++g)
        out[(size_t)(r0 + g) * kV + col] = acc[i][j][g] + bv;
    }
  }
}

// ---------------------------------------------------------------------------
// Fallback (no workspace): all-VGPR producer version.
// ---------------------------------------------------------------------------
__global__ __launch_bounds__(256, 1)
void rnnt_joiner_wmma_fb(const float* __restrict__ src,
                         const float* __restrict__ tgt,
                         const float* __restrict__ W,
                         const float* __restrict__ bias,
                         float* __restrict__ out)
{
  __shared__ _Float16 sA[2][BM * LDA];
  __shared__ _Float16 sB[2][BN * LDB];

  const int tid = threadIdx.x;
  const int bn0 = blockIdx.x * BN;
  const int bm0 = blockIdx.y * BM;

  const int a_c = (tid & 7) * 4;
  const int a_r = tid >> 3;
  int aSrcOff[4], aTgtOff[4], aRow[4];
#pragma unroll
  for (int p = 0; p < 4; ++p) {
    const int r    = a_r + p * 32;
    const int m    = bm0 + r;
    const int u    = m & (kU - 1);
    const int pidx = m >> 6;
    const int b    = pidx >> 7;
    aRow[p]    = r;
    aSrcOff[p] = pidx * kD;
    aTgtOff[p] = (b * kU + u) * kD;
  }
  const int b_v  = tid & 127;
  const int b_kg = tid >> 7;

  float4 ra[4], rb[4];
  auto load_tiles = [&](int ks) {
    const int k0 = ks * BK;
#pragma unroll
    for (int p = 0; p < 4; ++p) {
      const float4 s = *(const float4*)(src + aSrcOff[p] + k0 + a_c);
      const float4 t = *(const float4*)(tgt + aTgtOff[p] + k0 + a_c);
      float4 j;
      j.x = fmaxf(s.x + t.x, 0.f);
      j.y = fmaxf(s.y + t.y, 0.f);
      j.z = fmaxf(s.z + t.z, 0.f);
      j.w = fmaxf(s.w + t.w, 0.f);
      ra[p] = j;
    }
#pragma unroll
    for (int p = 0; p < 4; ++p) {
      const int kk = (b_kg + p * 2) * 4;
      const float* wp = W + (size_t)(k0 + kk) * kV + bn0 + b_v;
      float4 v;
      v.x = wp[0]; v.y = wp[kV]; v.z = wp[2 * kV]; v.w = wp[3 * kV];
      rb[p] = v;
    }
  };
  auto store_tiles = [&](int buf) {
#pragma unroll
    for (int p = 0; p < 4; ++p) {
      v4h h;
      h[0] = (_Float16)ra[p].x; h[1] = (_Float16)ra[p].y;
      h[2] = (_Float16)ra[p].z; h[3] = (_Float16)ra[p].w;
      *(v4h*)(&sA[buf][aRow[p] * LDA + a_c]) = h;
    }
#pragma unroll
    for (int p = 0; p < 4; ++p) {
      const int kk = (b_kg + p * 2) * 4;
      v4h h;
      h[0] = (_Float16)rb[p].x; h[1] = (_Float16)rb[p].y;
      h[2] = (_Float16)rb[p].z; h[3] = (_Float16)rb[p].w;
      *(v4h*)(&sB[buf][b_v * LDB + kk]) = h;
    }
  };

  const int lane  = tid & 31;
  const int w     = tid >> 5;
  const int wm    = (w & 3) * 32;
  const int wn    = (w >> 2) * 64;
  const int lrow  = lane & 15;
  const int khalf = (lane >> 4) * 8;

  v8f acc[2][4] = {};
  load_tiles(0);
#pragma unroll 2
  for (int ks = 0; ks < NK; ++ks) {
    const int buf = ks & 1;
    store_tiles(buf);
    __syncthreads();
    if (ks + 1 < NK) load_tiles(ks + 1);

    v16h afr[2], bfr[4];
#pragma unroll
    for (int i = 0; i < 2; ++i) {
      const _Float16* pa = &sA[buf][(wm + i * 16 + lrow) * LDA + khalf];
      const v8h lo = *(const v8h*)pa;
      const v8h hi = *(const v8h*)(pa + 16);
#pragma unroll
      for (int e = 0; e < 8; ++e) { afr[i][e] = lo[e]; afr[i][e + 8] = hi[e]; }
    }
#pragma unroll
    for (int j = 0; j < 4; ++j) {
      const _Float16* pb = &sB[buf][(wn + j * 16 + lrow) * LDB + khalf];
      const v8h lo = *(const v8h*)pb;
      const v8h hi = *(const v8h*)(pb + 16);
#pragma unroll
      for (int e = 0; e < 8; ++e) { bfr[j][e] = lo[e]; bfr[j][e + 8] = hi[e]; }
    }
#pragma unroll
    for (int i = 0; i < 2; ++i)
#pragma unroll
      for (int j = 0; j < 4; ++j)
        acc[i][j] = __builtin_amdgcn_wmma_f32_16x16x32_f16(
            false, afr[i], false, bfr[j], (short)0, acc[i][j], false, false);
  }
#pragma unroll
  for (int j = 0; j < 4; ++j) {
    const int col = bn0 + wn + j * 16 + lrow;
    const float bv = bias[col];
#pragma unroll
    for (int i = 0; i < 2; ++i) {
      const int r0 = bm0 + wm + i * 16 + ((lane >> 4) << 3);
#pragma unroll
      for (int g = 0; g < 8; ++g)
        out[(size_t)(r0 + g) * kV + col] = acc[i][j][g] + bv;
    }
  }
}

extern "C" void kernel_launch(void* const* d_in, const int* in_sizes, int n_in,
                              void* d_out, int out_size, void* d_ws, size_t ws_size,
                              hipStream_t stream) {
  (void)in_sizes; (void)n_in; (void)out_size;
  const float* src  = (const float*)d_in[0];   // (B,T,D)
  const float* tgt  = (const float*)d_in[1];   // (B,U,D)
  const float* W    = (const float*)d_in[2];   // (D,V)
  const float* bias = (const float*)d_in[3];   // (V,)
  float* out = (float*)d_out;                  // (B,T,U,V) fp32

  const dim3 grid(kV / BN, (kB * kT * kU) / BM);   // (8, 512)
  const size_t wtBytes = (size_t)kD * kV * sizeof(_Float16);  // 2 MB

  if (ws_size >= wtBytes) {
    _Float16* Wt = (_Float16*)d_ws;
    transpose_w_f16<<<dim3(kV / 64, kD / 64), 256, 0, stream>>>(W, Wt);
    rnnt_joiner_wmma_tdm<<<grid, dim3(256), 0, stream>>>(src, tgt, Wt, bias, out);
  } else {
    rnnt_joiner_wmma_fb<<<grid, dim3(256), 0, stream>>>(src, tgt, W, bias, out);
  }
}
